// SwitchLayer_67156108640623
// MI455X (gfx1250) — compile-verified, matrix operand order checked by reading, not run on hardware
//
#include <hip/hip_runtime.h>

// ---------------- problem constants (from reference) ----------------
#define NE    8        // experts
#define NS    2        // sublayers
#define DD    1024     // embed dim
#define FF_   4096     // ffn dim
#define TT    4096     // tokens (B*T_SEQ)
#define CAP   1024     // per-expert capacity (T/E*2.0)
#define MP    1152     // padded capacity rows: 9 * 128 (>= CAP+1)

// ---------------- GEMM tiling ----------------
#define BM 128
#define BN 128
#define BK 32

typedef __attribute__((ext_vector_type(16))) __bf16        v16bf;
typedef __attribute__((ext_vector_type(8)))  float         v8f;
typedef __attribute__((ext_vector_type(8)))  unsigned int  v8u;

__device__ __forceinline__ unsigned short f2bf(float f) {
    unsigned int u = __float_as_uint(f);
    u += 0x7FFFu + ((u >> 16) & 1u);        // round-to-nearest-even
    return (unsigned short)(u >> 16);
}

__device__ __forceinline__ float wave_sum(float v) {
#pragma unroll
    for (int off = 16; off > 0; off >>= 1) v += __shfl_xor(v, off, 32);
    return v;
}

// LDS tile layout is "fragment-ready": [tile16][lane][v] dwords, so a lane's whole
// 16x32 bf16 fragment is 8 contiguous dwords -> two ds_read_b128.
// CDNA5 16-bit A/B fragment map: half = lane>>4; VGPR v<4 -> K = 2v + 8*half,
// v>=4 -> K = 16 + 2(v-4) + 8*half. Pair index kp = (v&3) | (half<<2) | ((v>=4)<<3).
// Writer applies the inverse: half=(kp>>2)&1, v=(kp&3)|((kp&8)?4:0).
__device__ __forceinline__ v16bf frag_ld(const unsigned int* ldsTiles, int tile, int lane) {
    const uint4* q = (const uint4*)(ldsTiles + tile * 256 + lane * 8);
    uint4 lo = q[0], hi = q[1];
    v8u d = { lo.x, lo.y, lo.z, lo.w, hi.x, hi.y, hi.z, hi.w };
    return __builtin_bit_cast(v16bf, d);
}

// ---------------- kernels ----------------

// zero buf + h (padded capacity buffers); everything else is fully overwritten
__global__ __launch_bounds__(256) void zero_kernel(float4* __restrict__ buf4,
                                                   float4* __restrict__ h4, long n4) {
    long i = (long)blockIdx.x * 256 + threadIdx.x;
    float4 z = make_float4(0.f, 0.f, 0.f, 0.f);
    if (i < n4) { buf4[i] = z; h4[i] = z; }
}

// routing: per-token affinities, argmax, softmax partials (one wave per token, 8 waves/block)
__global__ __launch_bounds__(256) void routing_kernel(const float* __restrict__ feats,
                                                      const float* __restrict__ cent,
                                                      int* __restrict__ assign,
                                                      float* __restrict__ blockP) {
    __shared__ float pw[8][NE];
    const int wid = threadIdx.x >> 5, lane = threadIdx.x & 31;
    const int t = blockIdx.x * 8 + wid;
    const float* x = feats + (long)t * DD;
    float acc[NE];
#pragma unroll
    for (int j = 0; j < NE; ++j) acc[j] = 0.f;
    for (int d = lane; d < DD; d += 32) {
        float xv = x[d];
#pragma unroll
        for (int j = 0; j < NE; ++j) acc[j] += xv * cent[j * DD + d];
    }
#pragma unroll
    for (int j = 0; j < NE; ++j) acc[j] = wave_sum(acc[j]);
    if (lane == 0) {
        int arg = 0; float best = acc[0];
#pragma unroll
        for (int j = 1; j < NE; ++j) if (acc[j] > best) { best = acc[j]; arg = j; }
        assign[t] = arg;
        float sum = 0.f, p[NE];
#pragma unroll
        for (int j = 0; j < NE; ++j) { p[j] = expf(acc[j] - best); sum += p[j]; }
        float inv = 1.f / sum;
#pragma unroll
        for (int j = 0; j < NE; ++j) pw[wid][j] = p[j] * inv;
    }
    __syncthreads();
    if (threadIdx.x < NE) {            // deterministic per-block reduction of softmax probs
        float s = 0.f;
        for (int w = 0; w < 8; ++w) s += pw[w][threadIdx.x];
        blockP[blockIdx.x * NE + threadIdx.x] = s;
    }
}

// single-block stable prefix: position of each token within its expert queue (token order),
// plus counts and balance loss.
__global__ __launch_bounds__(256) void scan_kernel(const int* __restrict__ assign,
                                                   int* __restrict__ slot,
                                                   int* __restrict__ keep,
                                                   const float* __restrict__ blockP,
                                                   int nBlocksP,
                                                   float* __restrict__ out_loss) {
    __shared__ int sc[256];
    __shared__ int totals[NE];
    __shared__ float psum[NE];
    const int tid = threadIdx.x;
    const int base = tid * 16;
    int la[16], c[NE], off[NE];
#pragma unroll
    for (int e = 0; e < NE; ++e) c[e] = 0;
    for (int i = 0; i < 16; ++i) { la[i] = assign[base + i]; c[la[i]]++; }
    for (int e = 0; e < NE; ++e) {
        sc[tid] = c[e];
        __syncthreads();
        for (int d = 1; d < 256; d <<= 1) {          // Hillis-Steele inclusive scan
            int v = (tid >= d) ? sc[tid - d] : 0;
            __syncthreads();
            sc[tid] += v;
            __syncthreads();
        }
        off[e] = sc[tid] - c[e];
        if (tid == 255) totals[e] = sc[255];
        __syncthreads();
    }
    for (int i = 0; i < 16; ++i) {
        int e = la[i];
        int p = off[e]++;
        int k = (p < CAP) ? 1 : 0;
        keep[base + i] = k;
        slot[base + i] = k ? p : CAP;
    }
    if (tid < NE) {                                   // deterministic loss reduction
        float s = 0.f;
        for (int b = 0; b < nBlocksP; ++b) s += blockP[b * NE + tid];
        psum[tid] = s;
    }
    __syncthreads();
    if (tid == 0) {
        float loss = 0.f;
        for (int e = 0; e < NE; ++e) {
            float f = (float)totals[e] / (float)TT + 1e-6f;
            loss += psum[e] * f;
        }
        *out_loss = loss * (float)NE;
    }
}

// scatter tokens into capacity buffers (buf and h get the same data)
__global__ __launch_bounds__(256) void scatter_kernel(const float4* __restrict__ feats4,
                                                      const int* __restrict__ assign,
                                                      const int* __restrict__ slot,
                                                      float4* __restrict__ buf4,
                                                      float4* __restrict__ h4) {
    const int t = blockIdx.x;
    const int e = assign[t], sl = slot[t];
    long dst = ((long)e * MP + sl) * (DD / 4) + threadIdx.x;
    float4 v = feats4[(long)t * (DD / 4) + threadIdx.x];
    buf4[dst] = v;
    h4[dst] = v;
}

// alpha = softmax(buf @ cent^T)[row, own_expert]   (one wave per row)
__global__ __launch_bounds__(256) void alpha_kernel(const float* __restrict__ buf,
                                                    const float* __restrict__ cent,
                                                    float* __restrict__ alpha) {
    const int wid = threadIdx.x >> 5, lane = threadIdx.x & 31;
    const int row = blockIdx.x * 8 + wid;           // row in [0, NE*MP)
    const int e = row / MP;
    const float* x = buf + (long)row * DD;
    float acc[NE];
#pragma unroll
    for (int j = 0; j < NE; ++j) acc[j] = 0.f;
    for (int d = lane; d < DD; d += 32) {
        float xv = x[d];
#pragma unroll
        for (int j = 0; j < NE; ++j) acc[j] += xv * cent[j * DD + d];
    }
#pragma unroll
    for (int j = 0; j < NE; ++j) acc[j] = wave_sum(acc[j]);
    if (lane == 0) {
        float mx = acc[0];
#pragma unroll
        for (int j = 1; j < NE; ++j) mx = fmaxf(mx, acc[j]);
        float sum = 0.f;
#pragma unroll
        for (int j = 0; j < NE; ++j) sum += expf(acc[j] - mx);
        alpha[row] = expf(acc[e] - mx) / sum;
    }
}

// layernorm of h -> bf16 hn   (one wave per row)
__global__ __launch_bounds__(256) void ln_kernel(const float* __restrict__ h,
                                                 const float* __restrict__ ln_g,
                                                 const float* __restrict__ ln_b,
                                                 unsigned short* __restrict__ hn, int s) {
    const int wid = threadIdx.x >> 5, lane = threadIdx.x & 31;
    const int row = blockIdx.x * 8 + wid;
    const int e = row / MP;
    const float* x = h + (long)row * DD;
    float sum = 0.f, sq = 0.f;
    for (int d = lane; d < DD; d += 32) { float v = x[d]; sum += v; sq += v * v; }
    sum = wave_sum(sum); sq = wave_sum(sq);
    float mu = sum * (1.f / DD);
    float var = sq * (1.f / DD) - mu * mu;
    float rs = rsqrtf(var + 1e-5f);
    const float* g  = ln_g + (long)(s * NE + e) * DD;
    const float* bb = ln_b + (long)(s * NE + e) * DD;
    unsigned short* o = hn + (long)row * DD;
    for (int d = lane; d < DD; d += 32)
        o[d] = f2bf((x[d] - mu) * rs * g[d] + bb[d]);
}

// Tiled WMMA GEMM (double-buffered LDS, frag-ready layout, running-pointer addressing).
//   A : [NE][MP][KDIM] bf16 (activations)
//   W : [NE][KDIM][NDIM] fp32 (weights for this sublayer; converted to bf16 on the fly)
//   FFN1: out = relu(A@W + bias) -> bf16 outBf
//   else: outF += A@W + bias     (residual accumulate, fp32)
template <int KDIM, int NDIM, bool FFN1>
__global__ __launch_bounds__(256) void gemm_kernel(const unsigned short* __restrict__ A,
                                                   const float* __restrict__ W,
                                                   const float* __restrict__ bias,
                                                   unsigned short* __restrict__ outBf,
                                                   float* __restrict__ outF) {
    // frag-ready tiles: [8 tiles][32 lanes][8 dwords] = 2048 dwords = 8 KB per buffer
    __shared__ unsigned int lds_a[2][8 * 32 * 8];
    __shared__ unsigned int lds_b[2][8 * 32 * 8];

    const int e   = blockIdx.z;
    const int gm0 = blockIdx.x * BM;
    const int gn0 = blockIdx.y * BN;
    const int tid = threadIdx.x;
    const int lane = tid & 31, wid = tid >> 5;
    const int waveM = wid & 3, waveN = wid >> 2;    // 4x2 wave grid; wave tile 32x64
    const int half = lane >> 4, l16 = lane & 15;

    const float* bp = bias + (long)e * NDIM;

    // ---- running global pointers (all per-i / per-kt deltas are constants) ----
    // A: thread covers (m = (tid>>4) + 16*i, pair p = tid&15); per-kt advance BK/2 dwords.
    const unsigned int* aPtr = (const unsigned int*)(A + (long)e * MP * KDIM)
                             + (long)(gm0 + (tid >> 4)) * (KDIM / 2) + (tid & 15);
    // B: thread covers (kp = (tid>>7) + 2*i, n = tid&127); per-kt advance BK*NDIM floats.
    const float* bPtr = W + (long)e * KDIM * NDIM
                      + (long)((tid >> 7) * 2) * NDIM + gn0 + (tid & 127);

    // ---- constant LDS store offsets ----
    // A: m>>4 == i (tid < 256), so offset = i*256 + cA.
    const int pA  = tid & 15;
    const int hfA = (pA >> 2) & 1;
    const int vA  = (pA & 3) | ((pA & 8) ? 4 : 0);
    const int cA  = ((((tid >> 4) & 15) + (hfA << 4)) << 3) + vA;
    // B: offset = baseB + (hf_i<<7) + v_i, with kp_i = (tid>>7) + 2*i.
    const int nB    = tid & 127;
    const int baseB = ((nB >> 4) << 8) + ((nB & 15) << 3);
    const int kpB0  = tid >> 7;

    v8f acc[2][4];
    v8f vz = {0.f, 0.f, 0.f, 0.f, 0.f, 0.f, 0.f, 0.f};
#pragma unroll
    for (int i = 0; i < 2; ++i)
#pragma unroll
        for (int j = 0; j < 4; ++j) acc[i][j] = vz;

    unsigned int aReg[8];   // staged A pair-dwords for next tile
    unsigned int bReg[8];   // staged (converted) B pair-dwords for next tile

    auto loadA = [&]() {
#pragma unroll
        for (int i = 0; i < 8; ++i)
            aReg[i] = aPtr[(long)i * 8 * KDIM];     // delta = 16 rows * KDIM/2 dwords
        aPtr += BK / 2;
    };
    auto loadB = [&]() {
#pragma unroll
        for (int i = 0; i < 8; ++i) {
            float f0 = bPtr[(long)i * 4 * NDIM];     // kp advances by 2 -> k0 by 4
            float f1 = bPtr[(long)i * 4 * NDIM + NDIM];
            bReg[i] = (unsigned int)f2bf(f0) | ((unsigned int)f2bf(f1) << 16);
        }
        bPtr += (long)BK * NDIM;
    };
    auto storeA = [&](int bufi) {
#pragma unroll
        for (int i = 0; i < 8; ++i)
            lds_a[bufi][i * 256 + cA] = aReg[i];
    };
    auto storeB = [&](int bufi) {
#pragma unroll
        for (int i = 0; i < 8; ++i) {
            int kp = kpB0 + 2 * i;
            int hf = (kp >> 2) & 1;
            int v  = (kp & 3) | ((kp & 8) ? 4 : 0);
            lds_b[bufi][baseB + (hf << 7) + v] = bReg[i];
        }
    };

    constexpr int KT = KDIM / BK;
    loadA(); loadB();
    storeA(0); storeB(0);
    __syncthreads();

    for (int kt = 0; kt < KT; ++kt) {
        const int cur = kt & 1;
        if (kt + 1 < KT) { loadA(); loadB(); }       // prefetch next tile

        v16bf afr[2], bfr[4];
#pragma unroll
        for (int i = 0; i < 2; ++i)
            afr[i] = frag_ld(lds_a[cur], waveM * 2 + i, lane);
#pragma unroll
        for (int j = 0; j < 4; ++j)
            bfr[j] = frag_ld(lds_b[cur], waveN * 4 + j, lane);

#pragma unroll
        for (int i = 0; i < 2; ++i)
#pragma unroll
            for (int j = 0; j < 4; ++j)
                acc[i][j] = __builtin_amdgcn_wmma_f32_16x16x32_bf16(
                    false, afr[i], false, bfr[j], (short)0, acc[i][j], false, false);

        if (kt + 1 < KT) {
            storeA(cur ^ 1); storeB(cur ^ 1);
            __syncthreads();   // one barrier per K-step; also fences reads of `cur`
        }
    }

    // epilogue: C/D layout -> m = m0 + r + 8*half, n = n0 + l16
#pragma unroll
    for (int i = 0; i < 2; ++i) {
#pragma unroll
        for (int j = 0; j < 4; ++j) {
            int m0 = gm0 + waveM * 32 + i * 16;
            int n  = gn0 + waveN * 64 + j * 16 + l16;
            float bv = bp[n];
#pragma unroll
            for (int r = 0; r < 8; ++r) {
                int m = m0 + r + 8 * half;
                float v = acc[i][j][r] + bv;
                if constexpr (FFN1) {
                    v = v > 0.f ? v : 0.f;
                    outBf[((long)e * MP + m) * NDIM + n] = f2bf(v);
                } else {
                    long idx = ((long)e * MP + m) * NDIM + n;
                    outF[idx] = outF[idx] + v;        // residual: h = h + ff@W2 + b2
                }
            }
        }
    }
}

// combine: kept tokens get alpha-blended expert output, overflow passes through
__global__ __launch_bounds__(256) void combine_kernel(const float4* __restrict__ feats4,
                                                      const int* __restrict__ assign,
                                                      const int* __restrict__ slot,
                                                      const int* __restrict__ keep,
                                                      const float4* __restrict__ buf4,
                                                      const float4* __restrict__ h4,
                                                      const float* __restrict__ alpha,
                                                      float4* __restrict__ out4) {
    const int t = blockIdx.x;
    float4 r;
    if (keep[t]) {
        int e = assign[t], sl = slot[t];
        float a = alpha[(long)e * MP + sl];
        long src = ((long)e * MP + sl) * (DD / 4) + threadIdx.x;
        float4 hv = h4[src], bv = buf4[src];
        float om = 1.f - a;
        r.x = a * hv.x + om * bv.x;
        r.y = a * hv.y + om * bv.y;
        r.z = a * hv.z + om * bv.z;
        r.w = a * hv.w + om * bv.w;
    } else {
        r = feats4[(long)t * (DD / 4) + threadIdx.x];
    }
    out4[(long)t * (DD / 4) + threadIdx.x] = r;
}

// ---------------- host launcher ----------------
extern "C" void kernel_launch(void* const* d_in, const int* in_sizes, int n_in,
                              void* d_out, int out_size, void* d_ws, size_t ws_size,
                              hipStream_t stream) {
    const float* feats = (const float*)d_in[0];   // [T, D]
    const float* cent  = (const float*)d_in[1];   // [E, D]
    const float* W1    = (const float*)d_in[2];   // [S, E, D, F]
    const float* b1    = (const float*)d_in[3];   // [S, E, F]
    const float* W2    = (const float*)d_in[4];   // [S, E, F, D]
    const float* b2    = (const float*)d_in[5];   // [S, E, D]
    const float* ln_g  = (const float*)d_in[6];   // [S, E, D]
    const float* ln_b  = (const float*)d_in[7];   // [S, E, D]
    float* out = (float*)d_out;                   // [T*D] output + [1] loss

    // workspace carve-up (~170 MB total)
    char* ws = (char*)d_ws;
    float* buf = (float*)ws;                 ws += (long)NE * MP * DD * 4;     // 37.7 MB
    float* h   = (float*)ws;                 ws += (long)NE * MP * DD * 4;     // 37.7 MB
    unsigned short* ff = (unsigned short*)ws; ws += (long)NE * MP * FF_ * 2;   // 75.5 MB
    unsigned short* hn = (unsigned short*)ws; ws += (long)NE * MP * DD * 2;    // 18.9 MB
    float* alpha  = (float*)ws;              ws += (long)NE * MP * 4;
    float* blockP = (float*)ws;              ws += (long)(TT / 8) * NE * 4;
    int* assign = (int*)ws;                  ws += (long)TT * 4;
    int* slot   = (int*)ws;                  ws += (long)TT * 4;
    int* keep   = (int*)ws;                  ws += (long)TT * 4;

    const long n4 = (long)NE * MP * DD / 4;
    zero_kernel<<<dim3((unsigned)((n4 + 255) / 256)), 256, 0, stream>>>((float4*)buf, (float4*)h, n4);
    routing_kernel<<<dim3(TT / 8), 256, 0, stream>>>(feats, cent, assign, blockP);
    scan_kernel<<<dim3(1), 256, 0, stream>>>(assign, slot, keep, blockP, TT / 8, out + (long)TT * DD);
    scatter_kernel<<<dim3(TT), 256, 0, stream>>>((const float4*)feats, assign, slot,
                                                 (float4*)buf, (float4*)h);
    alpha_kernel<<<dim3(NE * MP / 8), 256, 0, stream>>>(buf, cent, alpha);

    for (int s = 0; s < NS; ++s) {
        ln_kernel<<<dim3(NE * MP / 8), 256, 0, stream>>>(h, ln_g, ln_b, hn, s);
        gemm_kernel<DD, FF_, true><<<dim3(MP / BM, FF_ / BN, NE), 256, 0, stream>>>(
            hn, W1 + (long)s * NE * DD * FF_, b1 + (long)s * NE * FF_, ff, nullptr);
        gemm_kernel<FF_, DD, false><<<dim3(MP / BM, DD / BN, NE), 256, 0, stream>>>(
            ff, W2 + (long)s * NE * FF_ * DD, b2 + (long)s * NE * DD, nullptr, h);
    }

    combine_kernel<<<dim3(TT), 256, 0, stream>>>((const float4*)feats, assign, slot, keep,
                                                 (const float4*)buf, (const float4*)h, alpha,
                                                 (float4*)out);
}
